// MultiHeadAttention_63118839382052
// MI455X (gfx1250) — matrix operand order repaired and checked by reading, not compiled, not run on hardware
//
#include <hip/hip_runtime.h>
#include <hip/hip_bf16.h>

typedef _Float16 v16h __attribute__((ext_vector_type(16)));
typedef _Float16 v8h  __attribute__((ext_vector_type(8)));
typedef _Float16 v4h  __attribute__((ext_vector_type(4)));
typedef float    v8f  __attribute__((ext_vector_type(8)));
typedef float    v4f  __attribute__((ext_vector_type(4)));

#define WMMA_F16(A, B, Cacc) \
  __builtin_amdgcn_wmma_f32_16x16x32_f16(false, (A), false, (B), (short)0, (Cacc), false, false)

// ---------------------------------------------------------------------------
// Fragment loaders (CDNA5 WMMA 16x16x32 f16, wave32)
// A 16x32: lane L -> row M = L&15, K runs [ko,ko+8) and [16+ko,16+ko+8), ko=(L>>4)*8
// B 32x16: lane L -> col N = L&15, K runs [ko,ko+8) and [16+ko,16+ko+8), ko=(L>>4)*8
// C/D 16x16 f32: vgpr i, lane L -> row = i + (L>>4)*8, col = L&15
// ---------------------------------------------------------------------------
static __device__ __forceinline__ v16h load_frag_a(const _Float16* base, int ld, int lane) {
  int r  = lane & 15;
  int ko = (lane >> 4) << 3;           // 0 or 8
  const _Float16* p = base + (long)r * ld + ko;
  v8h lo = *(const v8h*)(p);           // K = ko .. ko+7
  v8h hi = *(const v8h*)(p + 16);      // K = 16+ko .. 16+ko+7
  v16h out;
#pragma unroll
  for (int i = 0; i < 8; ++i) { out[i] = lo[i]; out[8 + i] = hi[i]; }
  return out;
}

static __device__ __forceinline__ v16h load_frag_b(const _Float16* base, int ld, int lane) {
  int n  = lane & 15;
  int ko = (lane >> 4) << 3;           // 0 or 8
  const _Float16* p = base + (long)n * ld + ko;
  v8h lo = *(const v8h*)(p);           // K = ko .. ko+7
  v8h hi = *(const v8h*)(p + 16);      // K = 16+ko .. 16+ko+7
  v16h out;
#pragma unroll
  for (int i = 0; i < 8; ++i) { out[i] = lo[i]; out[8 + i] = hi[i]; }
  return out;
}

// global_prefetch_b8 hints (probe-confirmed lowering on gfx1250)
static __device__ __forceinline__ void prefetch_row(const _Float16* base, int ld, int lane) {
  __builtin_prefetch((const void*)(base + (long)(lane & 15) * ld), 0, 1);
}

// ---------------------------------------------------------------------------
// f32 -> f16 conversion (x4 vectorized), grid-stride
// ---------------------------------------------------------------------------
__global__ void cvt_f32_to_f16_kernel(const v4f* __restrict__ in,
                                      v4h* __restrict__ out, int n4) {
  int i  = blockIdx.x * blockDim.x + threadIdx.x;
  int st = gridDim.x * blockDim.x;
  for (; i < n4; i += st) out[i] = __builtin_convertvector(in[i], v4h);
}

// ---------------------------------------------------------------------------
// QKV projection: out = x @ W.T (f16 in, f32 accum) + fused RoPE + layout.
//   mat 0 -> Q (B,H,T,D), RoPE, *1/sqrt(D)
//   mat 1 -> K (B,H,T,D), RoPE
//   mat 2 -> V transposed (B,H,D,T)
// Block = 128 threads (4 waves), tile 64 rows x 64 cols (one head).
// Ping-pong double buffer (unroll-2): no register rotation copies; each
// WMMA waits on loads issued a full iteration earlier.
// ---------------------------------------------------------------------------
__global__ __launch_bounds__(128)
void qkv_rope_kernel(const _Float16* __restrict__ xh,
                     const _Float16* __restrict__ wqh,
                     const _Float16* __restrict__ wkh,
                     const _Float16* __restrict__ wvh,
                     const float* __restrict__ cosp,
                     const float* __restrict__ sinp,
                     _Float16* __restrict__ qb,
                     _Float16* __restrict__ kb,
                     _Float16* __restrict__ vtb) {
  const int C = 1024, T = 2048, D = 64, H = 16;
  int lane = threadIdx.x & 31;
  int w    = threadIdx.x >> 5;
  int m0   = blockIdx.x * 64 + w * 16;  // token-row strip
  int n0   = blockIdx.y * 64;           // channel block == head*64
  int mat  = blockIdx.z;
  const _Float16* wh = (mat == 0) ? wqh : (mat == 1) ? wkh : wvh;
  const _Float16* abase = xh + (long)m0 * C;

  v8f acc[4] = {};
  v16h a0 = load_frag_a(abase + 0,  C, lane);
  v16h a1 = load_frag_a(abase + 32, C, lane);
  v16h b0[4], b1[4];
#pragma unroll
  for (int g = 0; g < 4; ++g) {
    b0[g] = load_frag_b(wh + (long)(n0 + g * 16) * C + 0,  C, lane);
    b1[g] = load_frag_b(wh + (long)(n0 + g * 16) * C + 32, C, lane);
  }

  for (int kk = 0; kk < C; kk += 64) {
    // ---- phase 0: consume (a0,b0) = data kk; refill with kk+64 ----
#pragma unroll
    for (int g = 0; g < 4; ++g) acc[g] = WMMA_F16(a0, b0[g], acc[g]);
    if (kk + 64 < C) {
      a0 = load_frag_a(abase + kk + 64, C, lane);
#pragma unroll
      for (int g = 0; g < 4; ++g)
        b0[g] = load_frag_b(wh + (long)(n0 + g * 16) * C + kk + 64, C, lane);
      prefetch_row(abase + kk + 256 < abase + C ? abase + kk + 256 : abase, C, lane);
      prefetch_row(wh + (long)n0 * C + ((kk + 256 < C) ? kk + 256 : 0), C, lane);
    }
    // ---- phase 1: consume (a1,b1) = data kk+32; refill with kk+96 ----
#pragma unroll
    for (int g = 0; g < 4; ++g) acc[g] = WMMA_F16(a1, b1[g], acc[g]);
    if (kk + 96 < C) {
      a1 = load_frag_a(abase + kk + 96, C, lane);
#pragma unroll
      for (int g = 0; g < 4; ++g)
        b1[g] = load_frag_b(wh + (long)(n0 + g * 16) * C + kk + 96, C, lane);
    }
  }

  int col  = lane & 15;
  int half = lane >> 4;
  int h    = blockIdx.y;
#pragma unroll
  for (int i = 0; i < 8; ++i) {
    int m    = m0 + half * 8 + i;
    int bidx = m >> 11;                 // / T
    int t    = m & (T - 1);
    if (mat < 2) {
      _Float16* dst = (mat == 0) ? qb : kb;
      long base = (((long)bidx * H + h) * T + t) * D;
      float qs = (mat == 0) ? 0.125f : 1.0f;   // fold 1/sqrt(64) into Q
#pragma unroll
      for (int g = 0; g < 2; ++g) {
        int d0  = g * 16 + col;          // [0,32)
        float c = cosp[t * 32 + d0];
        float s = sinp[t * 32 + d0];
        float x1 = acc[g][i], x2 = acc[g + 2][i];
        dst[base + d0]      = (_Float16)((x1 * c - x2 * s) * qs);
        dst[base + d0 + 32] = (_Float16)((x1 * s + x2 * c) * qs);
      }
    } else {
#pragma unroll
      for (int g = 0; g < 4; ++g) {
        int d = g * 16 + col;
        vtb[(((long)bidx * H + h) * D + d) * T + t] = (_Float16)acc[g][i];
      }
    }
  }
}

// ---------------------------------------------------------------------------
// Flash attention (causal, online softmax).
// Grid: (T/64 q-tiles, B*H). Block: 128 threads (4 waves), wave = 16 queries.
// Key-block count (2*qt+2) is even -> exact 2-step ping-pong on K fragments.
// Per phase: V loads issued first, score WMMAs consume the K buffer, buffer
// is refilled immediately (overlaps softmax + PV WMMAs).
// ---------------------------------------------------------------------------
__global__ __launch_bounds__(128)
void flash_attn_kernel(const _Float16* __restrict__ qb,
                       const _Float16* __restrict__ kb,
                       const _Float16* __restrict__ vtb,
                       _Float16* __restrict__ ob) {  // (B*T, C) row-major
  const int T = 2048, D = 64, H = 16, C = 1024;
  __shared__ __align__(16) _Float16 pbuf[4][16 * 32];

  int lane = threadIdx.x & 31;
  int w    = threadIdx.x >> 5;
  int qt   = blockIdx.x;
  int b    = blockIdx.y >> 4;
  int h    = blockIdx.y & 15;
  long hb  = (long)b * H + h;
  const _Float16* qbase = qb  + hb * T * D;
  const _Float16* kbase = kb  + hb * T * D;
  const _Float16* vbase = vtb + hb * D * T;

  int r0 = qt * 64 + w * 16;                       // this wave's query rows
  v16h qa0 = load_frag_a(qbase + (long)r0 * D + 0,  D, lane);
  v16h qa1 = load_frag_a(qbase + (long)r0 * D + 32, D, lane);

  v8f   o[4] = {};
  float mrun[8], lrun[8];
#pragma unroll
  for (int i = 0; i < 8; ++i) { mrun[i] = -3.0e38f; lrun[i] = 0.0f; }

  int col = lane & 15, half = lane >> 4;
  int nblocks = 2 * qt + 2;                        // even; keys up to qt*64+63

  // preload K fragments for blocks 0 and 1 (32 keys each)
  v16h kbuf0[4], kbuf1[4];
#pragma unroll
  for (int c = 0; c < 2; ++c) {
    kbuf0[2 * c + 0] = load_frag_b(kbase + (long)(0)       * D + 32 * c, D, lane);
    kbuf0[2 * c + 1] = load_frag_b(kbase + (long)(16)      * D + 32 * c, D, lane);
    kbuf1[2 * c + 0] = load_frag_b(kbase + (long)(32)      * D + 32 * c, D, lane);
    kbuf1[2 * c + 1] = load_frag_b(kbase + (long)(32 + 16) * D + 32 * c, D, lane);
  }
  // kbuf[2c+0] = keys g=0 (cols 0..15), d chunk c; kbuf[2c+1] = keys g=1, d chunk c

  auto phase = [&](int j, v16h (&kc)[4]) {
    int k0 = j * 32;
    // V fragments for this block: issued before anything consumes them
    v16h vblk[4];
#pragma unroll
    for (int g = 0; g < 4; ++g)
      vblk[g] = load_frag_b(vbase + (long)(g * 16) * T + k0, T, lane);

    v8f s0 = {}, s1 = {};
    s0 = WMMA_F16(qa0, kc[0], s0);
    s0 = WMMA_F16(qa1, kc[2], s0);
    s1 = WMMA_F16(qa0, kc[1], s1);
    s1 = WMMA_F16(qa1, kc[3], s1);

    // refill this K buffer with block j+2 (clamped); overlaps softmax + PV
    int kr = (j + 2 < nblocks) ? (j + 2) * 32 : 0;
#pragma unroll
    for (int c = 0; c < 2; ++c) {
      kc[2 * c + 0] = load_frag_b(kbase + (long)(kr)      * D + 32 * c, D, lane);
      kc[2 * c + 1] = load_frag_b(kbase + (long)(kr + 16) * D + 32 * c, D, lane);
    }

    int tk0 = k0 + col, tk1 = k0 + 16 + col;
#pragma unroll
    for (int i = 0; i < 8; ++i) {
      int tq   = r0 + half * 8 + i;
      float e0 = (tk0 <= tq) ? s0[i] : -3.0e38f;   // causal mask
      float e1 = (tk1 <= tq) ? s1[i] : -3.0e38f;
      float mx = fmaxf(e0, e1);
#pragma unroll
      for (int off = 8; off >= 1; off >>= 1)       // row max over 16-lane half
        mx = fmaxf(mx, __shfl_xor(mx, off, 32));
      float mnew = fmaxf(mrun[i], mx);
      float p0 = __expf(e0 - mnew);
      float p1 = __expf(e1 - mnew);
      float rs = p0 + p1;
#pragma unroll
      for (int off = 8; off >= 1; off >>= 1)
        rs += __shfl_xor(rs, off, 32);
      float scale = __expf(mrun[i] - mnew);
      mrun[i] = mnew;
      lrun[i] = lrun[i] * scale + rs;
#pragma unroll
      for (int g = 0; g < 4; ++g) o[g][i] *= scale;
      int row = half * 8 + i;                      // C-layout -> LDS tile
      pbuf[w][row * 32 + col]      = (_Float16)p0;
      pbuf[w][row * 32 + col + 16] = (_Float16)p1;
    }
    v16h pa = load_frag_a(&pbuf[w][0], 32, lane);  // A-layout reload (same wave)
#pragma unroll
    for (int g = 0; g < 4; ++g) o[g] = WMMA_F16(pa, vblk[g], o[g]);
  };

  for (int j = 0; j < nblocks; j += 2) {
    phase(j,     kbuf0);
    phase(j + 1, kbuf1);
  }

#pragma unroll
  for (int i = 0; i < 8; ++i) {
    int tq   = r0 + half * 8 + i;
    long m   = (long)b * T + tq;
    float inv = 1.0f / lrun[i];
#pragma unroll
    for (int g = 0; g < 4; ++g)
      ob[m * C + h * D + g * 16 + col] = (_Float16)(o[g][i] * inv);
  }
}

// ---------------------------------------------------------------------------
// Output projection: d_out = attn_out @ Wo.T (f32 result), ping-pong buffered.
// ---------------------------------------------------------------------------
__global__ __launch_bounds__(128)
void oproj_kernel(const _Float16* __restrict__ ah,
                  const _Float16* __restrict__ woh,
                  float* __restrict__ out) {
  const int C = 1024;
  int lane = threadIdx.x & 31;
  int w    = threadIdx.x >> 5;
  int m0   = blockIdx.x * 64 + w * 16;
  int n0   = blockIdx.y * 64;
  const _Float16* abase = ah + (long)m0 * C;

  v8f acc[4] = {};
  v16h a0 = load_frag_a(abase + 0,  C, lane);
  v16h a1 = load_frag_a(abase + 32, C, lane);
  v16h b0[4], b1[4];
#pragma unroll
  for (int g = 0; g < 4; ++g) {
    b0[g] = load_frag_b(woh + (long)(n0 + g * 16) * C + 0,  C, lane);
    b1[g] = load_frag_b(woh + (long)(n0 + g * 16) * C + 32, C, lane);
  }

  for (int kk = 0; kk < C; kk += 64) {
#pragma unroll
    for (int g = 0; g < 4; ++g) acc[g] = WMMA_F16(a0, b0[g], acc[g]);
    if (kk + 64 < C) {
      a0 = load_frag_a(abase + kk + 64, C, lane);
#pragma unroll
      for (int g = 0; g < 4; ++g)
        b0[g] = load_frag_b(woh + (long)(n0 + g * 16) * C + kk + 64, C, lane);
      prefetch_row(abase + ((kk + 256 < C) ? kk + 256 : 0), C, lane);
      prefetch_row(woh + (long)n0 * C + ((kk + 256 < C) ? kk + 256 : 0), C, lane);
    }
#pragma unroll
    for (int g = 0; g < 4; ++g) acc[g] = WMMA_F16(a1, b1[g], acc[g]);
    if (kk + 96 < C) {
      a1 = load_frag_a(abase + kk + 96, C, lane);
#pragma unroll
      for (int g = 0; g < 4; ++g)
        b1[g] = load_frag_b(woh + (long)(n0 + g * 16) * C + kk + 96, C, lane);
    }
  }

  int col = lane & 15, half = lane >> 4;
#pragma unroll
  for (int i = 0; i < 8; ++i) {
    long m = m0 + half * 8 + i;
#pragma unroll
    for (int g = 0; g < 4; ++g)
      out[m * C + n0 + g * 16 + col] = acc[g][i];
  }
}

// ---------------------------------------------------------------------------
// Launch: convert -> QKV+RoPE -> flash attention -> output projection
// Workspace layout (f16 unless noted), 48 MB total:
//   [0,8)MB xh | [8,10) Wq | [10,12) Wk | [12,14) Wv | [14,16) Wo
//   [16,24) Q(B,H,T,D) | [24,32) K(B,H,T,D) | [32,40) V^T(B,H,D,T)
//   [40,48) attn_out (B*T,C)
// ---------------------------------------------------------------------------
extern "C" void kernel_launch(void* const* d_in, const int* in_sizes, int n_in,
                              void* d_out, int out_size, void* d_ws, size_t ws_size,
                              hipStream_t stream) {
  (void)in_sizes; (void)n_in; (void)out_size; (void)ws_size;
  const float* x    = (const float*)d_in[0];
  const float* Wq   = (const float*)d_in[1];
  const float* Wk   = (const float*)d_in[2];
  const float* Wv   = (const float*)d_in[3];
  const float* Wo   = (const float*)d_in[4];
  const float* cosp = (const float*)d_in[5];
  const float* sinp = (const float*)d_in[6];
  // d_in[7] = causal mask, implicit in the flash kernel.

  const long MB = 1 << 20;
  char* ws = (char*)d_ws;
  _Float16* xh  = (_Float16*)(ws + 0 * MB);
  _Float16* wqh = (_Float16*)(ws + 8 * MB);
  _Float16* wkh = (_Float16*)(ws + 10 * MB);
  _Float16* wvh = (_Float16*)(ws + 12 * MB);
  _Float16* woh = (_Float16*)(ws + 14 * MB);
  _Float16* qbf = (_Float16*)(ws + 16 * MB);
  _Float16* kbf = (_Float16*)(ws + 24 * MB);
  _Float16* vtb = (_Float16*)(ws + 32 * MB);
  _Float16* aob = (_Float16*)(ws + 40 * MB);
  float* out = (float*)d_out;

  const int NX4 = (4096 * 1024) / 4;   // x elements / 4
  const int NW4 = (1024 * 1024) / 4;   // weight elements / 4
  cvt_f32_to_f16_kernel<<<2048, 256, 0, stream>>>((const v4f*)x,  (v4h*)xh,  NX4);
  cvt_f32_to_f16_kernel<<<1024, 256, 0, stream>>>((const v4f*)Wq, (v4h*)wqh, NW4);
  cvt_f32_to_f16_kernel<<<1024, 256, 0, stream>>>((const v4f*)Wk, (v4h*)wkh, NW4);
  cvt_f32_to_f16_kernel<<<1024, 256, 0, stream>>>((const v4f*)Wv, (v4h*)wvh, NW4);
  cvt_f32_to_f16_kernel<<<1024, 256, 0, stream>>>((const v4f*)Wo, (v4h*)woh, NW4);

  qkv_rope_kernel<<<dim3(64, 16, 3), 128, 0, stream>>>(
      xh, wqh, wkh, wvh, cosp, sinp, qbf, kbf, vtb);

  flash_attn_kernel<<<dim3(32, 32), 128, 0, stream>>>(qbf, kbf, vtb, aob);

  oproj_kernel<<<dim3(64, 16), 128, 0, stream>>>(aob, woh, out);
}
